// GraphClassifier_16621523435857
// MI455X (gfx1250) — matrix operand order repaired and checked by reading, not compiled, not run on hardware
//
#include <hip/hip_runtime.h>
#include <hip/hip_bf16.h>
#include <math.h>

// ---------------------------------------------------------------------------
// Problem constants (from the reference)
// ---------------------------------------------------------------------------
#define NN      20000        // nodes
#define EDGES   300000       // edges (before self loops)
#define ME      (EDGES + NN) // edges incl. self loops
#define BB      64           // graphs in batch
#define DIN     64
#define HID     64
#define HEADS   4
#define CC      256          // HID*HEADS
#define NCLS    10
#define LRELU   0.2f

typedef __attribute__((ext_vector_type(16))) _Float16 v16h;
typedef __attribute__((ext_vector_type(8)))  float    v8f;

// ---------------------------------------------------------------------------
// small helpers
// ---------------------------------------------------------------------------
__device__ __forceinline__ void atomicMaxF(float* addr, float val) {
    // IEEE bit-trick float atomic-max (correct for mixed signs)
    if (val >= 0.0f) {
        atomicMax((int*)addr, __float_as_int(val));
    } else {
        atomicMin((unsigned int*)addr, (unsigned int)__float_as_int(val));
    }
}

__device__ __forceinline__ float sigf(float x) { return 1.0f / (1.0f + expf(-x)); }

__global__ void k_fill(float* __restrict__ p, int n, float v) {
    int t = blockIdx.x * blockDim.x + threadIdx.x;
    if (t < n) p[t] = v;
}

__global__ void k_cvt_f16(const float* __restrict__ s, _Float16* __restrict__ d, int n) {
    int t = blockIdx.x * blockDim.x + threadIdx.x;
    if (t < n) d[t] = (_Float16)s[t];
}

// ---------------------------------------------------------------------------
// Pack W[K x 256] (f32, row-major) into per-lane WMMA B fragments (f16):
//   Bp[((tn*KB + kb)*32 + lane)*16 + i]  =  (f16) W[k*256 + col]
// where hi=lane>>4, col=tn*16+(lane&15), k=kb*32+hi*16+i.
// This makes the GEMM B-fragment two contiguous b128 loads per lane.
// ---------------------------------------------------------------------------
template <int K>
__global__ void k_packB(const float* __restrict__ W, _Float16* __restrict__ Bp) {
    constexpr int KB = K / 32;
    int t = blockIdx.x * blockDim.x + threadIdx.x;
    if (t >= 16 * KB * 32 * 16) return;
    int i    = t & 15;
    int lane = (t >> 4) & 31;
    int rest = t >> 9;
    int kb   = rest % KB;
    int tn   = rest / KB;
    int hi   = lane >> 4;
    int col  = tn * 16 + (lane & 15);
    int k    = kb * 32 + hi * 16 + i;
    Bp[t] = (_Float16)W[(size_t)k * CC + col];
}

// ---------------------------------------------------------------------------
// WMMA GEMM:  C[M x 256] = A[M x K] (f16) @ W (pre-packed B fragments), f32 acc.
// One wave per 16x16 output tile; K compile-time so the loop is a straight
// chain of v_wmma_f32_16x16x32_f16. C/D layout per CDNA5 ISA 7.12.2; A/B use contiguous k = 16*hi + i.
// ---------------------------------------------------------------------------
template <int K>
__global__ void k_gemm_wmma(const _Float16* __restrict__ A,
                            const _Float16* __restrict__ Bp,
                            float* __restrict__ Cout,
                            int rowTiles) {
    constexpr int KB = K / 32;
    const int wave  = blockIdx.x * (blockDim.x >> 5) + (threadIdx.x >> 5);
    const int total = rowTiles * (CC / 16);
    if (wave >= total) return;                 // wave-uniform
    const int tm   = wave >> 4;
    const int tn   = wave & 15;
    const int lane = threadIdx.x & 31;
    const int hi   = lane >> 4;                // lane half-group (0 / 1)
    const int l15  = lane & 15;

    const int row = tm * 16 + l15;             // A row for this lane
    const int col = tn * 16 + l15;             // C column for this lane

    v8f acc = {0.f, 0.f, 0.f, 0.f, 0.f, 0.f, 0.f, 0.f};

#pragma unroll
    for (int kb = 0; kb < KB; ++kb) {
        const int k0 = kb * 32;
        // ---- A fragment: 16x32 f16 (contiguous k = 16*hi + i, matching packed B)
        union { v16h v; uint4 u[2]; } af;
        const _Float16* ap = A + (size_t)row * K + k0 + (hi ? 16 : 0);
        af.u[0] = *reinterpret_cast<const uint4*>(ap);
        af.u[1] = *reinterpret_cast<const uint4*>(ap + 8);

        // ---- B fragment: pre-packed, 32 contiguous bytes per lane
        union { v16h v; uint4 u[2]; } bf;
        const _Float16* bp = Bp + ((size_t)(tn * KB + kb) * 32 + lane) * 16;
        bf.u[0] = *reinterpret_cast<const uint4*>(bp);
        bf.u[1] = *reinterpret_cast<const uint4*>(bp + 8);

        acc = __builtin_amdgcn_wmma_f32_16x16x32_f16(
            false, af.v, false, bf.v, (short)0, acc, false, false);
    }

    // ---- C/D 16x16 f32: lane<16 -> M = 0..7, lane>=16 -> M = 8..15
    const int r0 = tm * 16 + hi * 8;
#pragma unroll
    for (int v = 0; v < 8; ++v)
        Cout[(size_t)(r0 + v) * CC + col] = acc[v];
}

// ---------------------------------------------------------------------------
// GAT attention pieces
// ---------------------------------------------------------------------------
// al[n][h] = sum_f feat[n][h*64+f]*a_s[h][f] ;  ar likewise with a_d
__global__ void k_alar(const float* __restrict__ feat,
                       const float* __restrict__ as_, const float* __restrict__ ad_,
                       float* __restrict__ al, float* __restrict__ ar) {
    int t = blockIdx.x * blockDim.x + threadIdx.x;
    if (t >= NN * HEADS) return;
    int n = t >> 2, h = t & 3;
    const float* f = feat + (size_t)n * CC + h * HID;
    const float* s = as_ + h * HID;
    const float* d = ad_ + h * HID;
    float sl = 0.f, sr = 0.f;
#pragma unroll 4
    for (int i = 0; i < HID; ++i) { float x = f[i]; sl += x * s[i]; sr += x * d[i]; }
    al[t] = sl; ar[t] = sr;
}

__device__ __forceinline__ void edge_sd(const int* ei, int e, int& s, int& d) {
    if (e < EDGES) { s = ei[e]; d = ei[EDGES + e]; }
    else           { s = e - EDGES; d = s; }        // self loop
}

__global__ void k_edge_max(const int* __restrict__ ei,
                           const float* __restrict__ al, const float* __restrict__ ar,
                           float* __restrict__ emax) {
    int t = blockIdx.x * blockDim.x + threadIdx.x;
    if (t >= ME * HEADS) return;
    int e = t >> 2, h = t & 3, s, d;
    edge_sd(ei, e, s, d);
    float x = al[s * 4 + h] + ar[d * 4 + h];
    x = x > 0.f ? x : LRELU * x;
    atomicMaxF(&emax[d * 4 + h], x);
}

__global__ void k_edge_pden(const int* __restrict__ ei,
                            const float* __restrict__ al, const float* __restrict__ ar,
                            const float* __restrict__ emax,
                            float* __restrict__ pbuf, float* __restrict__ den) {
    int t = blockIdx.x * blockDim.x + threadIdx.x;
    if (t >= ME * HEADS) return;
    int e = t >> 2, h = t & 3, s, d;
    edge_sd(ei, e, s, d);
    float x = al[s * 4 + h] + ar[d * 4 + h];
    x = x > 0.f ? x : LRELU * x;
    float em = emax[d * 4 + h];
    if (em < -1e29f) em = 0.f;
    float p = expf(x - em);
    pbuf[t] = p;
    atomicAdd(&den[d * 4 + h], p);
}

// acc[d][c] += p * feat[s][c] ; one thread handles 8 channels of one head
__global__ void k_edge_agg(const int* __restrict__ ei,
                           const float* __restrict__ pbuf,
                           const float* __restrict__ feat,
                           float* __restrict__ acc) {
    int t = blockIdx.x * blockDim.x + threadIdx.x;
    if (t >= ME * 32) return;
    int e = t >> 5, sub = t & 31;
    int h = sub >> 3, grp = sub & 7, s, d;
    edge_sd(ei, e, s, d);
    int cb = h * HID + grp * 8;
    const float* fs = feat + (size_t)s * CC + cb;
    __builtin_prefetch(fs, 0, 1);              // global_prefetch_b8 ahead of gather
    float p = pbuf[e * 4 + h];
    float* ad = acc + (size_t)d * CC + cb;
#pragma unroll
    for (int j = 0; j < 8; ++j) atomicAdd(ad + j, p * fs[j]);
}

// h_out = relu(acc/den + bias) (+ residual) ; also emit f16 copy for next GEMM
__global__ void k_finish(const float* __restrict__ acc, const float* __restrict__ den,
                         const float* __restrict__ bias,
                         float* __restrict__ hio, _Float16* __restrict__ xh,
                         int residual) {
    int t = blockIdx.x * blockDim.x + threadIdx.x;
    if (t >= NN * CC) return;
    int n = t >> 8, c = t & 255, h = c >> 6;
    float v = acc[t] / den[n * 4 + h] + bias[c];
    v = v > 0.f ? v : 0.f;
    if (residual) v += hio[t];
    hio[t] = v;
    xh[t]  = (_Float16)v;
}

// ---------------------------------------------------------------------------
// Set2Set pooling (B=64, C=256, LSTM(2C->C), torch gate order i,f,g,o)
// ---------------------------------------------------------------------------
__global__ void k_gates(const float* __restrict__ qstar, const float* __restrict__ hs,
                        const float* __restrict__ Wih, const float* __restrict__ Whh,
                        const float* __restrict__ bih, const float* __restrict__ bhh,
                        float* __restrict__ gates) {
    int t = blockIdx.x * blockDim.x + threadIdx.x;       // 64 * 1024
    if (t >= BB * 4 * CC) return;
    int b = t >> 10, g = t & 1023;
    const float* q  = qstar + (size_t)b * 2 * CC;
    const float* hh = hs + (size_t)b * CC;
    const float* wi = Wih + (size_t)g * 2 * CC;
    const float* wh = Whh + (size_t)g * CC;
    float s = bih[g] + bhh[g];
#pragma unroll 4
    for (int k = 0; k < 2 * CC; ++k) s += q[k] * wi[k];
#pragma unroll 4
    for (int k = 0; k < CC; ++k) s += hh[k] * wh[k];
    gates[t] = s;
}

__global__ void k_lstm(const float* __restrict__ gates,
                       float* __restrict__ hs, float* __restrict__ cs,
                       float* __restrict__ qstar) {
    int t = blockIdx.x * blockDim.x + threadIdx.x;       // 64 * 256
    if (t >= BB * CC) return;
    int b = t >> 8, c = t & 255;
    const float* g = gates + (size_t)b * 4 * CC;
    float i  = sigf(g[c]);
    float f  = sigf(g[CC + c]);
    float gg = tanhf(g[2 * CC + c]);
    float o  = sigf(g[3 * CC + c]);
    float cn = f * cs[t] + i * gg;
    float hn = o * tanhf(cn);
    cs[t] = cn; hs[t] = hn;
    qstar[(size_t)b * 2 * CC + c]      = hn;   // q half for next iteration
    qstar[(size_t)b * 2 * CC + CC + c] = 0.f;  // reset r half (accumulated later)
}

__global__ void k_s2s_e(const float* __restrict__ h, const float* __restrict__ q,
                        const int* __restrict__ batch,
                        float* __restrict__ e_n, float* __restrict__ emaxB) {
    int n = blockIdx.x * blockDim.x + threadIdx.x;
    if (n >= NN) return;
    int b = batch[n];
    const float* hp = h + (size_t)n * CC;
    const float* qp = q + (size_t)b * CC;
    float s = 0.f;
#pragma unroll 4
    for (int i = 0; i < CC; ++i) s += hp[i] * qp[i];
    e_n[n] = s;
    atomicMaxF(&emaxB[b], s);
}

__global__ void k_s2s_pden(const float* __restrict__ e_n, const float* __restrict__ emaxB,
                           const int* __restrict__ batch,
                           float* __restrict__ p_n, float* __restrict__ denB) {
    int n = blockIdx.x * blockDim.x + threadIdx.x;
    if (n >= NN) return;
    int b = batch[n];
    float em = emaxB[b];
    if (em < -1e29f) em = 0.f;
    float p = expf(e_n[n] - em);
    p_n[n] = p;
    atomicAdd(&denB[b], p);
}

// r accumulation into qstar[:, C:2C] : one thread = (node, 8 channels)
__global__ void k_s2s_r(const float* __restrict__ p_n, const float* __restrict__ denB,
                        const int* __restrict__ batch, const float* __restrict__ h,
                        float* __restrict__ qstar) {
    int t = blockIdx.x * blockDim.x + threadIdx.x;
    if (t >= NN * 32) return;
    int n = t >> 5, grp = t & 31;
    int b = batch[n];
    float a = p_n[n] / (denB[b] + 1e-16f);
    int cb = grp * 8;
    const float* hp = h + (size_t)n * CC + cb;
    float* rp = qstar + (size_t)b * 2 * CC + CC + cb;
#pragma unroll
    for (int j = 0; j < 8; ++j) atomicAdd(rp + j, a * hp[j]);
}

// ---------------------------------------------------------------------------
// Classifier head
// ---------------------------------------------------------------------------
__global__ void k_cls1(const float* __restrict__ qstar, const float* __restrict__ Wc1,
                       const float* __restrict__ bc1, float* __restrict__ z) {
    int t = blockIdx.x * blockDim.x + threadIdx.x;       // 64 * 64
    if (t >= BB * HID) return;
    int b = t >> 6, j = t & 63;
    const float* q = qstar + (size_t)b * 2 * CC;
    float s = bc1[j];
#pragma unroll 4
    for (int k = 0; k < 2 * CC; ++k) s += q[k] * Wc1[(size_t)k * HID + j];
    z[t] = s > 0.f ? s : 0.f;
}

__global__ void k_cls2(const float* __restrict__ z, const float* __restrict__ Wc2,
                       const float* __restrict__ bc2, float* __restrict__ out) {
    int t = blockIdx.x * blockDim.x + threadIdx.x;       // 64 * 10
    if (t >= BB * NCLS) return;
    int b = t / NCLS, c = t % NCLS;
    const float* zp = z + (size_t)b * HID;
    float s = bc2[c];
#pragma unroll
    for (int j = 0; j < HID; ++j) s += zp[j] * Wc2[(size_t)j * NCLS + c];
    out[t] = s;
}

// ---------------------------------------------------------------------------
// Host-side launcher
// ---------------------------------------------------------------------------
static inline unsigned gridFor(long n, int bs) { return (unsigned)((n + bs - 1) / bs); }
static inline size_t aUp(size_t x) { return (x + 255) & ~(size_t)255; }

extern "C" void kernel_launch(void* const* d_in, const int* in_sizes, int n_in,
                              void* d_out, int out_size, void* d_ws, size_t ws_size,
                              hipStream_t stream) {
    (void)in_sizes; (void)n_in; (void)out_size;
    const float* x     = (const float*)d_in[0];
    const int*   ei    = (const int*)  d_in[1];
    const int*   batch = (const int*)  d_in[2];
    const float* Wl[3]  = {(const float*)d_in[3], (const float*)d_in[7],  (const float*)d_in[11]};
    const float* asl[3] = {(const float*)d_in[4], (const float*)d_in[8],  (const float*)d_in[12]};
    const float* adl[3] = {(const float*)d_in[5], (const float*)d_in[9],  (const float*)d_in[13]};
    const float* bl[3]  = {(const float*)d_in[6], (const float*)d_in[10], (const float*)d_in[14]};
    const float* Wih = (const float*)d_in[15];
    const float* Whh = (const float*)d_in[16];
    const float* bih = (const float*)d_in[17];
    const float* bhh = (const float*)d_in[18];
    const float* Wc1 = (const float*)d_in[19];
    const float* bc1 = (const float*)d_in[20];
    const float* Wc2 = (const float*)d_in[21];
    const float* bc2 = (const float*)d_in[22];

    // ---- workspace partition (all 256B aligned)
    char* w = (char*)d_ws; size_t off = 0;
    auto grab = [&](size_t bytes) { char* p = w + off; off += aUp(bytes); return p; };
    _Float16* xh   = (_Float16*)grab((size_t)NN * CC * 2);
    _Float16* wpk  = (_Float16*)grab((size_t)CC * CC * 2);   // packed B fragments
    float* feat    = (float*)grab((size_t)NN * CC * 4);
    float* accb    = (float*)grab((size_t)NN * CC * 4);
    float* hbuf    = (float*)grab((size_t)NN * CC * 4);
    float* al      = (float*)grab((size_t)NN * HEADS * 4);
    float* ar      = (float*)grab((size_t)NN * HEADS * 4);
    float* emax    = (float*)grab((size_t)NN * HEADS * 4);
    float* den     = (float*)grab((size_t)NN * HEADS * 4);
    float* pbuf    = (float*)grab((size_t)ME * HEADS * 4);
    float* e_n     = (float*)grab((size_t)NN * 4);
    float* p_n     = (float*)grab((size_t)NN * 4);
    float* emaxB   = (float*)grab((size_t)BB * 4);
    float* denB    = (float*)grab((size_t)BB * 4);
    float* qstar   = (float*)grab((size_t)BB * 2 * CC * 4);
    float* hs      = (float*)grab((size_t)BB * CC * 4);
    float* cs      = (float*)grab((size_t)BB * CC * 4);
    float* gates   = (float*)grab((size_t)BB * 4 * CC * 4);
    float* zbuf    = (float*)grab((size_t)BB * HID * 4);
    if (off > ws_size) return;   // workspace too small: bail deterministically

    const int BS = 256;
    const int rowTiles = NN / 16;                    // 1250
    const unsigned gemmBlocks = gridFor((long)rowTiles * (CC / 16), BS / 32);

    // ======================= 3 GAT layers =======================
    for (int l = 0; l < 3; ++l) {
        const int K = (l == 0) ? DIN : CC;
        // pack W into WMMA B fragments (fused f32->f16 convert)
        if (K == DIN)
            k_packB<DIN><<<gridFor((long)16 * (DIN / 32) * 512, BS), BS, 0, stream>>>(Wl[l], wpk);
        else
            k_packB<CC><<<gridFor((long)16 * (CC / 32) * 512, BS), BS, 0, stream>>>(Wl[l], wpk);
        if (l == 0)
            k_cvt_f16<<<gridFor((long)NN * DIN, BS), BS, 0, stream>>>(x, xh, NN * DIN);
        // feat = X @ W  (WMMA f16 -> f32)
        if (K == DIN)
            k_gemm_wmma<DIN><<<gemmBlocks, BS, 0, stream>>>(xh, wpk, feat, rowTiles);
        else
            k_gemm_wmma<CC><<<gemmBlocks, BS, 0, stream>>>(xh, wpk, feat, rowTiles);
        // attention scalars
        k_alar<<<gridFor((long)NN * HEADS, BS), BS, 0, stream>>>(feat, asl[l], adl[l], al, ar);
        // init accumulators
        k_fill<<<gridFor((long)NN * HEADS, BS), BS, 0, stream>>>(emax, NN * HEADS, -1e30f);
        k_fill<<<gridFor((long)NN * HEADS, BS), BS, 0, stream>>>(den,  NN * HEADS, 0.f);
        k_fill<<<gridFor((long)NN * CC, BS), BS, 0, stream>>>(accb, NN * CC, 0.f);
        // softmax over incoming edges + aggregation
        k_edge_max <<<gridFor((long)ME * HEADS, BS), BS, 0, stream>>>(ei, al, ar, emax);
        k_edge_pden<<<gridFor((long)ME * HEADS, BS), BS, 0, stream>>>(ei, al, ar, emax, pbuf, den);
        k_edge_agg <<<gridFor((long)ME * 32, BS), BS, 0, stream>>>(ei, pbuf, feat, accb);
        // h = relu(acc/den + b) (+ residual for layers 1,2); also write f16 copy
        k_finish<<<gridFor((long)NN * CC, BS), BS, 0, stream>>>(accb, den, bl[l], hbuf, xh, l > 0);
    }

    // ======================= Set2Set (3 steps) =======================
    k_fill<<<gridFor((long)BB * 2 * CC, BS), BS, 0, stream>>>(qstar, BB * 2 * CC, 0.f);
    k_fill<<<gridFor((long)BB * CC, BS), BS, 0, stream>>>(hs, BB * CC, 0.f);
    k_fill<<<gridFor((long)BB * CC, BS), BS, 0, stream>>>(cs, BB * CC, 0.f);
    for (int s = 0; s < 3; ++s) {
        k_gates<<<gridFor((long)BB * 4 * CC, BS), BS, 0, stream>>>(qstar, hs, Wih, Whh, bih, bhh, gates);
        k_lstm <<<gridFor((long)BB * CC, BS), BS, 0, stream>>>(gates, hs, cs, qstar);
        k_fill <<<1, BB, 0, stream>>>(emaxB, BB, -1e30f);
        k_fill <<<1, BB, 0, stream>>>(denB,  BB, 0.f);
        k_s2s_e   <<<gridFor(NN, BS), BS, 0, stream>>>(hbuf, hs, batch, e_n, emaxB);
        k_s2s_pden<<<gridFor(NN, BS), BS, 0, stream>>>(e_n, emaxB, batch, p_n, denB);
        k_s2s_r   <<<gridFor((long)NN * 32, BS), BS, 0, stream>>>(p_n, denB, batch, hbuf, qstar);
    }

    // ======================= classifier head =======================
    k_cls1<<<gridFor((long)BB * HID, BS), BS, 0, stream>>>(qstar, Wc1, bc1, zbuf);
    k_cls2<<<gridFor((long)BB * NCLS, BS), BS, 0, stream>>>(zbuf, Wc2, bc2, (float*)d_out);
}
